// Fan_69295002354092
// MI455X (gfx1250) — matrix operand order repaired and checked by reading, not compiled, not run on hardware
//
#include <hip/hip_runtime.h>
#include <hip/hip_bf16.h>

#define BATCHN 64
#define SEQN   256
#define EMBEDN 300
#define EPAD   320
#define HIDN   512
#define G3     1536
#define D2     1024
#define NCLSN  10
#define NROWS  (BATCHN * SEQN)   // 16384

typedef __attribute__((ext_vector_type(16))) _Float16 v16h;
typedef __attribute__((ext_vector_type(8)))  _Float16 v8h;
typedef __attribute__((ext_vector_type(8)))  float    v8f;

union V16U { v16h v; v8h h[2]; };

// A fragment (16x32 f16): lane holds row m = lane&15; contiguous K = [16*hf, 16*hf+16)
__device__ __forceinline__ v16h load_fragA(const _Float16* base, int ld, int lane) {
    int m  = lane & 15;
    int hf = lane >> 4;
    V16U u;
    u.h[0] = *(const v8h*)(base + (long)m * ld + 16 * hf);
    u.h[1] = *(const v8h*)(base + (long)m * ld + 16 * hf + 8);
    return u.v;
}

// B fragment (32x16 f16) from row-major W[N,K]: lane holds col n = lane&15,
// contiguous K = [16*hf, 16*hf+16)
__device__ __forceinline__ v16h load_fragB(const _Float16* base, int ld, int lane) {
    int n  = lane & 15;
    int hf = lane >> 4;
    return *(const v16h*)(base + (long)n * ld + 16 * hf);
}

__device__ __forceinline__ v8f wmma16(v16h a, v16h b, v8f c) {
    return __builtin_amdgcn_wmma_f32_16x16x32_f16(false, a, false, b, (short)0, c, false, false);
}

__device__ __forceinline__ float sigmoidf_(float x) {
    return 1.f / (1.f + __expf(-x));
}

// Async-copy `bytes` (multiple of 16*blockDim.x not required; caller ensures chunk count
// divides evenly) from contiguous global memory into LDS via GLOBAL_LOAD_ASYNC_TO_LDS_B128
// (ASYNCcnt). ldsOff = workgroup-relative LDS byte offset (low 32 bits of shared pointer).
__device__ __forceinline__ void async_copy_to_lds(unsigned ldsOff, const void* gsrc,
                                                  int bytes, int tid, int nthreads) {
    int chunks = bytes >> 4;   // 16B per lane per op
    for (int it = tid; it < chunks; it += nthreads) {
        unsigned long long ga = (unsigned long long)((const char*)gsrc + (long)it * 16);
        unsigned la = ldsOff + (unsigned)it * 16u;
        asm volatile("global_load_async_to_lds_b128 %0, %1, off"
                     :: "v"(la), "v"(ga) : "memory");
    }
}

__device__ __forceinline__ void wait_async_all() {
    asm volatile("s_wait_asynccnt 0x0" ::: "memory");
}

// ---------------- 1) embedding gather + f16 convert, pad E 300 -> 320 ----------------
__global__ void k_embed(const int* __restrict__ tokens, const float* __restrict__ emb,
                        _Float16* __restrict__ X) {
    int r = blockIdx.x;            // r = s*64 + b
    int s = r >> 6, b = r & 63;
    int tok = tokens[b * SEQN + s];
    const float* src = emb + (long)tok * EMBEDN;
    for (int e = threadIdx.x; e < EPAD; e += blockDim.x)
        X[(long)r * EPAD + e] = (e < EMBEDN) ? (_Float16)src[e] : (_Float16)0.f;
}

// ---------------- 2) f32 -> f16 weight conversion with column padding ----------------
__global__ void k_cvt(const float* __restrict__ src, _Float16* __restrict__ dst,
                      int rows, int cols, int cpad) {
    long i = (long)blockIdx.x * blockDim.x + threadIdx.x;
    long tot = (long)rows * cpad;
    if (i >= tot) return;
    int rr = (int)(i / cpad), cc = (int)(i % cpad);
    dst[i] = (cc < cols) ? (_Float16)src[(long)rr * cols + cc] : (_Float16)0.f;
}

// ---------------- 3) gx = X @ w_ih^T  (store f16) ----------------
// grid (24, 1024, 2), block 128: A tile async-staged to LDS, wave -> one 16x16 tile
__global__ void k_gemm_gx(const _Float16* __restrict__ X, const _Float16* __restrict__ Wih,
                          _Float16* __restrict__ GX) {
    __shared__ _Float16 aT[16 * EPAD];    // 10 KB, contiguous 16-row A tile
    int tid  = threadIdx.x;
    int wave = tid >> 5, lane = tid & 31;
    int nt = blockIdx.x * 4 + wave;      // < 96
    int mt = blockIdx.y;                 // < 1024
    int d  = blockIdx.z;

    // stage A tile (contiguous 10240 B) into LDS with ASYNCcnt-tracked copies
    async_copy_to_lds((unsigned)(size_t)(void*)aT, X + (long)mt * 16 * EPAD,
                      16 * EPAD * 2, tid, blockDim.x);
    wait_async_all();
    __syncthreads();

    const _Float16* B0 = Wih + ((long)d * G3 + (long)nt * 16) * EPAD;
    v8f c = {};
    for (int k = 0; k < EPAD; k += 32) {
        __builtin_prefetch(B0 + k + 64, 0, 1);
        v16h a = load_fragA(aT + k, EPAD, lane);    // ds_load_b128
        v16h b = load_fragB(B0 + k, EPAD, lane);    // global_load_b128 (L2)
        c = wmma16(a, b, c);
    }
    _Float16* D = GX + ((long)d * NROWS + (long)mt * 16) * G3 + nt * 16;
    int n = lane & 15, hf = lane >> 4;
    for (int v = 0; v < 8; ++v)
        D[(long)(v + 8 * hf) * G3 + n] = (_Float16)c[v];
}

// ---------------- 4) persistent GRU scan: 2 blocks (one per dir), 1024 threads ----------------
__global__ void k_gru(const _Float16* __restrict__ GX, const _Float16* __restrict__ Whh,
                      const float* __restrict__ b_ih, const float* __restrict__ b_hh,
                      _Float16* __restrict__ Hout) {
    extern __shared__ char smem[];
    _Float16* hS  = (_Float16*)smem;                          // [64][512]  64 KB
    _Float16* ghS = (_Float16*)(smem + BATCHN * HIDN * 2);    // [64][1536] 192 KB
    int d = blockIdx.x;
    int tid = threadIdx.x;
    int wave = tid >> 5, lane = tid & 31;
    for (int i = tid; i < BATCHN * HIDN; i += blockDim.x) hS[i] = (_Float16)0.f;
    __syncthreads();

    const _Float16* W = Whh + (long)d * G3 * HIDN;
    const float* bih = b_ih + d * G3;
    const float* bhh = b_hh + d * G3;

    for (int s = 0; s < SEQN; ++s) {
        // gh = h @ W_hh^T : 4 mtiles x 96 ntiles = 384 tiles over 32 waves
        for (int t = wave; t < 4 * 96; t += 32) {
            int mt = t & 3, nt = t >> 2;
            const _Float16* A0 = hS + (long)mt * 16 * HIDN;
            const _Float16* B0 = W + (long)nt * 16 * HIDN;
            v8f c = {};
            for (int k = 0; k < HIDN; k += 32) {
                v16h a = load_fragA(A0 + k, HIDN, lane);   // ds_load from LDS
                v16h b = load_fragB(B0 + k, HIDN, lane);   // L2-resident weights
                c = wmma16(a, b, c);
            }
            int n = lane & 15, hf = lane >> 4;
            _Float16* Dp = ghS + (long)(mt * 16 + 8 * hf) * G3 + nt * 16 + n;
            for (int v = 0; v < 8; ++v) Dp[(long)v * G3] = (_Float16)c[v];
        }
        __syncthreads();

        // elementwise gate math: 64*512 elems / 1024 threads = 32 each (uniform)
        const _Float16* gxrow = GX + ((long)d * NROWS + (long)s * BATCHN) * G3;
        for (int i = tid; i < BATCHN * HIDN; i += blockDim.x) {
            int b = i >> 9, j = i & (HIDN - 1);
            float xr = (float)gxrow[(long)b * G3 + j]            + bih[j];
            float xz = (float)gxrow[(long)b * G3 + HIDN + j]     + bih[HIDN + j];
            float xn = (float)gxrow[(long)b * G3 + 2 * HIDN + j] + bih[2 * HIDN + j];
            float hr = (float)ghS[(long)b * G3 + j]              + bhh[j];
            float hz = (float)ghS[(long)b * G3 + HIDN + j]       + bhh[HIDN + j];
            float hn = (float)ghS[(long)b * G3 + 2 * HIDN + j]   + bhh[2 * HIDN + j];
            float r = sigmoidf_(xr + hr);
            float z = sigmoidf_(xz + hz);
            float nn = tanhf(xn + r * hn);
            float hnew = (1.f - z) * nn + z * (float)hS[i];
            hS[i] = (_Float16)hnew;
            Hout[((long)b * SEQN + s) * D2 + d * HIDN + j] = (_Float16)hnew;
        }
        __syncthreads();
    }
}

// ---------------- 5) attention scores: score[r] = ctx . tanh(h[r] @ attn_w^T + b) ----------------
// grid 1024 (one 16-row M tile each), block 128 (4 waves x 16 N tiles).
// A tile (32 KB, contiguous) async-staged into LDS once, reused by all 64 N tiles.
__global__ void k_attn(const _Float16* __restrict__ Hout, const _Float16* __restrict__ Waw,
                       const float* __restrict__ attn_b, const float* __restrict__ ctx,
                       float* __restrict__ scores) {
    __shared__ _Float16 aT[16 * D2];     // 32 KB
    __shared__ float ssc[16];
    int tid = threadIdx.x;
    int mt = blockIdx.x;
    int wave = tid >> 5, lane = tid & 31;
    if (tid < 16) ssc[tid] = 0.f;

    async_copy_to_lds((unsigned)(size_t)(void*)aT, Hout + (long)mt * 16 * D2,
                      16 * D2 * 2, tid, blockDim.x);
    wait_async_all();
    __syncthreads();

    int n = lane & 15, hf = lane >> 4;
    float acc[8] = {0.f, 0.f, 0.f, 0.f, 0.f, 0.f, 0.f, 0.f};
    for (int nt = wave; nt < 64; nt += 4) {
        const _Float16* B0 = Waw + (long)nt * 16 * D2;
        v8f c = {};
        for (int k = 0; k < D2; k += 32) {
            __builtin_prefetch(B0 + k + 64, 0, 1);
            v16h a = load_fragA(aT + k, D2, lane);     // ds_load_b128
            v16h b = load_fragB(B0 + k, D2, lane);     // global_load_b128 (L2)
            c = wmma16(a, b, c);
        }
        int ncol = nt * 16 + n;
        float bb = attn_b[ncol], cc = ctx[ncol];
        for (int v = 0; v < 8; ++v)
            acc[v] += tanhf(c[v] + bb) * cc;
    }
    for (int v = 0; v < 8; ++v)
        atomicAdd(&ssc[v + 8 * hf], acc[v]);     // ds_add_f32 reduction across lanes/waves
    __syncthreads();
    if (tid < 16) scores[(long)mt * 16 + tid] = ssc[tid];
}

// ---------------- 6) softmax over S, weighted sum, fc, log_softmax ----------------
__global__ void k_final(const _Float16* __restrict__ Hout, const float* __restrict__ scores,
                        const float* __restrict__ fc_w, const float* __restrict__ fc_b,
                        float* __restrict__ out) {
    __shared__ float alpha[SEQN];
    __shared__ float wsum[D2];
    __shared__ float red[SEQN];
    __shared__ float lg[NCLSN];
    int b = blockIdx.x, tid = threadIdx.x;   // 256 threads
    float sc = scores[(long)b * SEQN + tid];
    red[tid] = sc; __syncthreads();
    for (int off = 128; off; off >>= 1) {
        if (tid < off) red[tid] = fmaxf(red[tid], red[tid + off]);
        __syncthreads();
    }
    float mx = red[0]; __syncthreads();
    float e = __expf(sc - mx);
    red[tid] = e; __syncthreads();
    for (int off = 128; off; off >>= 1) {
        if (tid < off) red[tid] += red[tid + off];
        __syncthreads();
    }
    float inv = 1.f / red[0];
    alpha[tid] = e * inv;
    __syncthreads();
    for (int d0 = tid; d0 < D2; d0 += 256) {
        float acc = 0.f;
        const _Float16* hp = Hout + (long)b * SEQN * D2 + d0;
        for (int s = 0; s < SEQN; ++s) acc += alpha[s] * (float)hp[(long)s * D2];
        wsum[d0] = acc;
    }
    __syncthreads();
    if (tid < NCLSN) {
        float acc = fc_b[tid];
        const float* w = fc_w + (long)tid * D2;
        for (int j = 0; j < D2; ++j) acc += w[j] * wsum[j];
        lg[tid] = acc;
    }
    __syncthreads();
    if (tid == 0) {
        float m = lg[0];
        for (int c = 1; c < NCLSN; ++c) m = fmaxf(m, lg[c]);
        float ss = 0.f;
        for (int c = 0; c < NCLSN; ++c) ss += __expf(lg[c] - m);
        float lse = m + __logf(ss);
        for (int c = 0; c < NCLSN; ++c) out[b * NCLSN + c] = lg[c] - lse;
    }
}

extern "C" void kernel_launch(void* const* d_in, const int* in_sizes, int n_in,
                              void* d_out, int out_size, void* d_ws, size_t ws_size,
                              hipStream_t stream) {
    const int*   tokens = (const int*)  d_in[0];
    const float* emb    = (const float*)d_in[1];
    const float* w_ih   = (const float*)d_in[2];
    const float* w_hh   = (const float*)d_in[3];
    const float* b_ih   = (const float*)d_in[4];
    const float* b_hh   = (const float*)d_in[5];
    const float* attn_w = (const float*)d_in[6];
    const float* attn_b = (const float*)d_in[7];
    const float* ctx    = (const float*)d_in[8];
    const float* fc_w   = (const float*)d_in[9];
    const float* fc_b   = (const float*)d_in[10];
    float* out = (float*)d_out;

    char* ws = (char*)d_ws;
    size_t off = 0;
    auto alloc = [&](size_t bytes) {
        char* p = ws + off;
        off += (bytes + 255) & ~(size_t)255;
        return p;
    };
    _Float16* X     = (_Float16*)alloc((size_t)NROWS * EPAD * 2);         // 10.5 MB
    _Float16* Wih16 = (_Float16*)alloc((size_t)2 * G3 * EPAD * 2);        //  2.0 MB
    _Float16* Whh16 = (_Float16*)alloc((size_t)2 * G3 * HIDN * 2);        //  3.1 MB
    _Float16* Waw16 = (_Float16*)alloc((size_t)D2 * D2 * 2);              //  2.1 MB
    _Float16* GX    = (_Float16*)alloc((size_t)2 * NROWS * G3 * 2);       // 100.7 MB
    _Float16* Hout  = (_Float16*)alloc((size_t)NROWS * D2 * 2);           // 33.6 MB
    float*    Sc    = (float*)   alloc((size_t)NROWS * 4);                // 64 KB

    // 1) embedding gather + f16 conversion
    k_embed<<<NROWS, 128, 0, stream>>>(tokens, emb, X);

    // 2) weight conversions
    {
        long t1 = (long)2 * G3 * EPAD;
        k_cvt<<<(unsigned)((t1 + 255) / 256), 256, 0, stream>>>(w_ih, Wih16, 2 * G3, EMBEDN, EPAD);
        long t2 = (long)2 * G3 * HIDN;
        k_cvt<<<(unsigned)((t2 + 255) / 256), 256, 0, stream>>>(w_hh, Whh16, 2 * G3, HIDN, HIDN);
        long t3 = (long)D2 * D2;
        k_cvt<<<(unsigned)((t3 + 255) / 256), 256, 0, stream>>>(attn_w, Waw16, D2, D2, D2);
    }

    // 3) input projections (WMMA GEMM, async LDS staging)
    k_gemm_gx<<<dim3(24, NROWS / 16, 2), 128, 0, stream>>>(X, Wih16, GX);

    // 4) persistent GRU scan, one workgroup per direction, 256 KB dynamic LDS
    size_t ldsz = (size_t)BATCHN * HIDN * 2 + (size_t)BATCHN * G3 * 2;
    k_gru<<<2, 1024, ldsz, stream>>>(GX, Whh16, b_ih, b_hh, Hout);

    // 5) attention scores (WMMA GEMM + async LDS staging + fused tanh/ctx reduction)
    k_attn<<<NROWS / 16, 128, 0, stream>>>(Hout, Waw16, attn_b, ctx, Sc);

    // 6) softmax + weighted sum + fc + log_softmax
    k_final<<<BATCHN, 256, 0, stream>>>(Hout, Sc, fc_w, fc_b, out);
}
